// GCNLAYER_61589831024880
// MI455X (gfx1250) — matrix operand order repaired and checked by reading, not compiled, not run on hardware
//
#include <hip/hip_runtime.h>
#include <hip/hip_bf16.h>

typedef _Float16 f16;
typedef __attribute__((ext_vector_type(4)))  _Float16 v4h;
typedef __attribute__((ext_vector_type(8)))  _Float16 v8h;
typedef __attribute__((ext_vector_type(16))) _Float16 v16h;
typedef __attribute__((ext_vector_type(8)))  float    v8f;

// ---------------------------------------------------------------------------
// Utility kernels
// ---------------------------------------------------------------------------
__global__ __launch_bounds__(256) void k_fill0(float* __restrict__ p, long n) {
    long i = (long)blockIdx.x * 256 + threadIdx.x;
    if (i < n) p[i] = 0.0f;
}

// f32 -> f16, 4 elements per thread
__global__ __launch_bounds__(256) void k_cvt16(const float* __restrict__ x,
                                               f16* __restrict__ y, long n4) {
    long i = (long)blockIdx.x * 256 + threadIdx.x;
    if (i < n4) {
        const float4 v = *(const float4*)(x + i * 4);
        v4h o = {(f16)v.x, (f16)v.y, (f16)v.z, (f16)v.w};
        *(v4h*)(y + i * 4) = o;
    }
}

// Wt[n*128+k] = (f16) W[(k+k_off)*128 + n]  (transpose 128x128 chunk + convert)
__global__ __launch_bounds__(256) void k_wtcvt(const float* __restrict__ W,
                                               f16* __restrict__ Wt, int k_off) {
    int t = blockIdx.x * 256 + threadIdx.x;   // 16384 total
    int n = t >> 7, k = t & 127;
    Wt[t] = (f16)W[(long)(k + k_off) * 128 + n];
}

// per-node attention partial dots: pp[n*8+k]=h[n].Wa_k[0:128], pp[n*8+4+k]=h[n].Wa_k[128:256]
__global__ __launch_bounds__(256) void k_attp(const float* __restrict__ h,
                                              const float* __restrict__ Wa1,
                                              const float* __restrict__ Wa2,
                                              const float* __restrict__ Wa3,
                                              float* __restrict__ pp, int N) {
    int n = blockIdx.x * 256 + threadIdx.x;
    if (n >= N) return;
    const float* hr = h + (long)n * 128;
    float s1 = 0, s2 = 0, s3 = 0, d1 = 0, d2 = 0, d3 = 0;
    for (int c = 0; c < 128; c += 4) {
        float4 hv = *(const float4*)(hr + c);
        float4 w1a = *(const float4*)(Wa1 + c);
        float4 w2a = *(const float4*)(Wa2 + c);
        float4 w3a = *(const float4*)(Wa3 + c);
        float4 w1b = *(const float4*)(Wa1 + 128 + c);
        float4 w2b = *(const float4*)(Wa2 + 128 + c);
        float4 w3b = *(const float4*)(Wa3 + 128 + c);
        s1 += hv.x*w1a.x + hv.y*w1a.y + hv.z*w1a.z + hv.w*w1a.w;
        s2 += hv.x*w2a.x + hv.y*w2a.y + hv.z*w2a.z + hv.w*w2a.w;
        s3 += hv.x*w3a.x + hv.y*w3a.y + hv.z*w3a.z + hv.w*w3a.w;
        d1 += hv.x*w1b.x + hv.y*w1b.y + hv.z*w1b.z + hv.w*w1b.w;
        d2 += hv.x*w2b.x + hv.y*w2b.y + hv.z*w2b.z + hv.w*w2b.w;
        d3 += hv.x*w3b.x + hv.y*w3b.y + hv.z*w3b.z + hv.w*w3b.w;
    }
    float* o = pp + (long)n * 8;
    o[0] = s1; o[1] = s2; o[2] = s3; o[3] = 0.0f;
    o[4] = d1; o[5] = d2; o[6] = d3; o[7] = 0.0f;
}

// ---------------------------------------------------------------------------
// WMMA GEMM: C[M,128] = act(A[M,128] @ W + (A2 @ W2) + bias)
// Wt/Wt2 pre-transposed f16 (Wt[n][k] = W[k][n]); staged in LDS per block.
// NS strips of 16 rows per wave; n-tiles processed in pairs -> 2*NS
// independent WMMA accumulator chains share each set of B-fragment LDS loads.
// ---------------------------------------------------------------------------
__device__ __forceinline__ v16h ldfragA(const f16* p) {
    v8h lo = *(const v8h*)(p);
    v8h hi = *(const v8h*)(p + 16);
    return __builtin_shufflevector(lo, hi, 0,1,2,3,4,5,6,7,8,9,10,11,12,13,14,15);
}
__device__ __forceinline__ v16h ldfragB(const f16* p) {
    v8h lo = *(const v8h*)(p);
    v8h hi = *(const v8h*)(p + 16);
    return __builtin_shufflevector(lo, hi, 0,1,2,3,4,5,6,7,8,9,10,11,12,13,14,15);
}

template<bool DUAL, int NS>
__global__ __launch_bounds__(256) void k_gemm128(
    const f16* __restrict__ A, const f16* __restrict__ Wt,
    const float* __restrict__ bias,
    const f16* __restrict__ A2, const f16* __restrict__ Wt2,
    f16* __restrict__ out16, float* __restrict__ out32, int M, int do_relu)
{
    extern __shared__ f16 lB[];               // 16384 halves (+16384 if DUAL)

    // cooperative stage of weight matrices into LDS
    {
        const int t = threadIdx.x;
        #pragma unroll
        for (int i = 0; i < 8; i++) {
            int idx = t * 8 + i * 2048;
            *(v8h*)(lB + idx) = *(const v8h*)(Wt + idx);
        }
        if (DUAL) {
            #pragma unroll
            for (int i = 0; i < 8; i++) {
                int idx = t * 8 + i * 2048;
                *(v8h*)(lB + 16384 + idx) = *(const v8h*)(Wt2 + idx);
            }
        }
    }
    __syncthreads();

    const int lane = threadIdx.x & 31;
    const int wave = threadIdx.x >> 5;
    const int base = blockIdx.x * (128 * NS) + wave * (16 * NS);
    const int aoff = (lane >> 4) * 8;    // A-frag K sub-offset per lane half
    const int boff = (lane >> 4) * 8;    // B-frag K sub-offset per lane half
    const int ncol = lane & 15;

    v16h afr[NS][4], a2fr[NS][4];
    #pragma unroll
    for (int s = 0; s < NS; s++) {
        int mrow = base + s * 16 + (lane & 15);
        long mr = (long)(mrow < M ? mrow : M - 1);
        const f16* ap = A + mr * 128 + aoff;
        #pragma unroll
        for (int k = 0; k < 4; k++) afr[s][k] = ldfragA(ap + k * 32);
        if (DUAL) {
            const f16* ap2 = A2 + mr * 128 + aoff;
            #pragma unroll
            for (int k = 0; k < 4; k++) a2fr[s][k] = ldfragA(ap2 + k * 32);
        }
    }

    for (int n0 = 0; n0 < 8; n0 += 2) {
        v8f acc[NS][2];
        #pragma unroll
        for (int s = 0; s < NS; s++) {
            acc[s][0] = (v8f){0.f,0.f,0.f,0.f,0.f,0.f,0.f,0.f};
            acc[s][1] = (v8f){0.f,0.f,0.f,0.f,0.f,0.f,0.f,0.f};
        }
        const f16* bp0 = lB + (n0 * 16 + ncol) * 128 + boff;
        const f16* bp1 = bp0 + 16 * 128;
        v16h b0[4], b1[4];
        #pragma unroll
        for (int k = 0; k < 4; k++) {
            b0[k] = ldfragB(bp0 + k * 32);
            b1[k] = ldfragB(bp1 + k * 32);
        }
        #pragma unroll
        for (int k = 0; k < 4; k++) {
            #pragma unroll
            for (int s = 0; s < NS; s++) {
                acc[s][0] = __builtin_amdgcn_wmma_f32_16x16x32_f16(
                    false, afr[s][k], false, b0[k], (short)0, acc[s][0], false, false);
                acc[s][1] = __builtin_amdgcn_wmma_f32_16x16x32_f16(
                    false, afr[s][k], false, b1[k], (short)0, acc[s][1], false, false);
            }
        }
        if (DUAL) {
            const f16* cp0 = lB + 16384 + (n0 * 16 + ncol) * 128 + boff;
            const f16* cp1 = cp0 + 16 * 128;
            #pragma unroll
            for (int k = 0; k < 4; k++) {
                b0[k] = ldfragB(cp0 + k * 32);
                b1[k] = ldfragB(cp1 + k * 32);
            }
            #pragma unroll
            for (int k = 0; k < 4; k++) {
                #pragma unroll
                for (int s = 0; s < NS; s++) {
                    acc[s][0] = __builtin_amdgcn_wmma_f32_16x16x32_f16(
                        false, a2fr[s][k], false, b0[k], (short)0, acc[s][0], false, false);
                    acc[s][1] = __builtin_amdgcn_wmma_f32_16x16x32_f16(
                        false, a2fr[s][k], false, b1[k], (short)0, acc[s][1], false, false);
                }
            }
        }
        // epilogue
        #pragma unroll
        for (int s = 0; s < NS; s++) {
            const int rbase = base + s * 16 + ((lane >> 4) << 3);
            #pragma unroll
            for (int half = 0; half < 2; half++) {
                const v8f& a = acc[s][half];
                const int col = (n0 + half) * 16 + ncol;
                float bv = bias[col];
                float vv[8];
                #pragma unroll
                for (int r = 0; r < 8; r++) {
                    float v = a[r] + bv;
                    vv[r] = do_relu ? fmaxf(v, 0.0f) : v;
                }
                if (out16) {
                    #pragma unroll
                    for (int r = 0; r < 8; r++) {
                        int row = rbase + r;
                        if (row < M) out16[(long)row * 128 + col] = (f16)vv[r];
                    }
                }
                if (out32) {
                    #pragma unroll
                    for (int r = 0; r < 8; r++) {
                        int row = rbase + r;
                        if (row < M) out32[(long)row * 128 + col] = vv[r];
                    }
                }
            }
        }
    }
}

// ---------------------------------------------------------------------------
// Edge kernels
// ---------------------------------------------------------------------------
// accumulate per-destination attention normalizers + degree
__global__ __launch_bounds__(256) void k_edge_att(
    const int* __restrict__ src, const int* __restrict__ dst,
    const float* __restrict__ pp,
    const float* __restrict__ ba1, const float* __restrict__ ba2,
    const float* __restrict__ ba3, float* __restrict__ acc, int E)
{
    int e = blockIdx.x * 256 + threadIdx.x;
    if (e >= E) return;
    int s = src[e], d = dst[e];
    float4 ps = *(const float4*)(pp + (long)s * 8);
    float4 pd = *(const float4*)(pp + (long)d * 8 + 4);
    float a1 = __expf(fmaxf(ps.x + pd.x + ba1[0], 0.0f));
    float a2 = __expf(fmaxf(ps.y + pd.y + ba2[0], 0.0f));
    float a3 = __expf(fmaxf(ps.z + pd.z + ba3[0], 0.0f));
    float* ac = acc + (long)d * 4;
    atomicAdd(ac + 0, a1);
    atomicAdd(ac + 1, a2);
    atomicAdd(ac + 2, a3);
    atomicAdd(ac + 3, 1.0f);
}

// one wave per edge: lane handles 4 of the 128 message channels
__global__ __launch_bounds__(256) void k_edge_scatter(
    const int* __restrict__ src, const int* __restrict__ dst,
    const float* __restrict__ pp, const float* __restrict__ acc,
    const f16* __restrict__ msg16,
    const float* __restrict__ ba1, const float* __restrict__ ba2,
    const float* __restrict__ ba3, float* __restrict__ hneigh, int E)
{
    int lane = threadIdx.x & 31;
    int e = blockIdx.x * 8 + (threadIdx.x >> 5);
    if (e >= E) return;
    int s = src[e], d = dst[e];
    float4 ps = *(const float4*)(pp + (long)s * 8);
    float4 pd = *(const float4*)(pp + (long)d * 8 + 4);
    float4 at = *(const float4*)(acc + (long)d * 4);
    float a1 = __expf(fmaxf(ps.x + pd.x + ba1[0], 0.0f));
    float a2 = __expf(fmaxf(ps.y + pd.y + ba2[0], 0.0f));
    float a3 = __expf(fmaxf(ps.z + pd.z + ba3[0], 0.0f));
    float avg = (a1 / at.x + a2 / at.y + a3 / at.z) * (1.0f / 3.0f);
    v4h mv = *(const v4h*)(msg16 + (long)s * 128 + lane * 4);
    float* hp = hneigh + (long)d * 128 + lane * 4;
    atomicAdd(hp + 0, avg * (float)mv[0]);
    atomicAdd(hp + 1, avg * (float)mv[1]);
    atomicAdd(hp + 2, avg * (float)mv[2]);
    atomicAdd(hp + 3, avg * (float)mv[3]);
}

// hn16 = (f16)(hneigh / max(deg,1)), 4 channels per thread
__global__ __launch_bounds__(256) void k_norm(
    const float* __restrict__ hneigh, const float* __restrict__ acc,
    f16* __restrict__ hn16, int N)
{
    long t = (long)blockIdx.x * 256 + threadIdx.x;
    int node = (int)(t >> 5);
    int c4 = (int)(t & 31) * 4;
    if (node >= N) return;
    float inv = 1.0f / fmaxf(acc[(long)node * 4 + 3], 1.0f);
    float4 v = *(const float4*)(hneigh + (long)node * 128 + c4);
    v4h o = {(f16)(v.x * inv), (f16)(v.y * inv), (f16)(v.z * inv), (f16)(v.w * inv)};
    *(v4h*)(hn16 + (long)node * 128 + c4) = o;
}

// ---------------------------------------------------------------------------
// Launch
// ---------------------------------------------------------------------------
extern "C" void kernel_launch(void* const* d_in, const int* in_sizes, int n_in,
                              void* d_out, int out_size, void* d_ws, size_t ws_size,
                              hipStream_t stream) {
    (void)n_in; (void)out_size; (void)ws_size;
    const float* h    = (const float*)d_in[0];
    const int*   src  = (const int*)d_in[1];
    const int*   dst  = (const int*)d_in[2];
    const float* Wm1  = (const float*)d_in[3];  const float* bm1 = (const float*)d_in[4];
    const float* Wm2  = (const float*)d_in[5];  const float* bm2 = (const float*)d_in[6];
    const float* Wm3  = (const float*)d_in[7];  const float* bm3 = (const float*)d_in[8];
    const float* Wa1  = (const float*)d_in[9];  const float* ba1 = (const float*)d_in[10];
    const float* Wa2  = (const float*)d_in[11]; const float* ba2 = (const float*)d_in[12];
    const float* Wa3  = (const float*)d_in[13]; const float* ba3 = (const float*)d_in[14];
    const float* Wc1  = (const float*)d_in[15]; const float* bc1 = (const float*)d_in[16];
    const float* Wc2  = (const float*)d_in[17]; const float* bc2 = (const float*)d_in[18];

    const int N = in_sizes[0] / 128;
    const int E = in_sizes[1];

    // workspace carve-out (~80 MB for N=50000)
    char* ws = (char*)d_ws;
    auto carve = [&](size_t bytes) -> char* {
        char* p = ws;
        ws += (bytes + 255) & ~(size_t)255;
        return p;
    };
    f16*   hh     = (f16*)  carve((size_t)N * 128 * 2);
    f16*   tA     = (f16*)  carve((size_t)N * 128 * 2);   // MLP L1 out, later hn16
    f16*   tB     = (f16*)  carve((size_t)N * 128 * 2);   // MLP L2 out, later cls hidden
    f16*   msg16  = (f16*)  carve((size_t)N * 128 * 2);   // MLP L3 out (node messages)
    float* hneigh = (float*)carve((size_t)N * 128 * 4);
    float* pp     = (float*)carve((size_t)N * 8 * 4);
    float* acc    = (float*)carve((size_t)N * 4 * 4);     // att1,att2,att3,deg per node
    f16*   W1t    = (f16*)  carve(128 * 128 * 2);
    f16*   W2t    = (f16*)  carve(128 * 128 * 2);
    f16*   W3t    = (f16*)  carve(128 * 128 * 2);
    f16*   Wc1at  = (f16*)  carve(128 * 128 * 2);
    f16*   Wc1bt  = (f16*)  carve(128 * 128 * 2);
    f16*   Wc2t   = (f16*)  carve(128 * 128 * 2);

    // 1. zero accumulators
    {
        long nz = (long)N * 128;
        k_fill0<<<(unsigned)((nz + 255) / 256), 256, 0, stream>>>(hneigh, nz);
        long nz2 = (long)N * 4;
        k_fill0<<<(unsigned)((nz2 + 255) / 256), 256, 0, stream>>>(acc, nz2);
    }
    // 2. convert node features to f16
    {
        long n4 = (long)N * 32;
        k_cvt16<<<(unsigned)((n4 + 255) / 256), 256, 0, stream>>>(h, hh, n4);
    }
    // 3. transpose+convert weights
    k_wtcvt<<<64, 256, 0, stream>>>(Wm1, W1t, 0);
    k_wtcvt<<<64, 256, 0, stream>>>(Wm2, W2t, 0);
    k_wtcvt<<<64, 256, 0, stream>>>(Wm3, W3t, 0);
    k_wtcvt<<<64, 256, 0, stream>>>(Wc1, Wc1at, 0);
    k_wtcvt<<<64, 256, 0, stream>>>(Wc1, Wc1bt, 128);
    k_wtcvt<<<64, 256, 0, stream>>>(Wc2, Wc2t, 0);
    // 4. per-node attention partial dots
    k_attp<<<(unsigned)((N + 255) / 256), 256, 0, stream>>>(h, Wa1, Wa2, Wa3, pp, N);
    // 5-7. node-level message MLP (WMMA), msg = relu(relu(relu(h W1+b1) W2+b2) W3+b3)
    const unsigned gblk1 = (unsigned)((N + 255) / 256);   // NS=2: 256 rows/block
    const unsigned gblk2 = (unsigned)((N + 127) / 128);   // NS=1: 128 rows/block
    const size_t shm1 = 16384 * sizeof(f16);      // one weight matrix in LDS
    const size_t shm2 = 2 * 16384 * sizeof(f16);  // two weight matrices in LDS
    k_gemm128<false,2><<<gblk1, 256, shm1, stream>>>(hh, W1t, bm1, nullptr, nullptr, tA,    nullptr, N, 1);
    k_gemm128<false,2><<<gblk1, 256, shm1, stream>>>(tA, W2t, bm2, nullptr, nullptr, tB,    nullptr, N, 1);
    k_gemm128<false,2><<<gblk1, 256, shm1, stream>>>(tB, W3t, bm3, nullptr, nullptr, msg16, nullptr, N, 1);
    // 8. per-edge attention normalizer + degree accumulation
    k_edge_att<<<(unsigned)((E + 255) / 256), 256, 0, stream>>>(
        src, dst, pp, ba1, ba2, ba3, acc, E);
    // 9. wave-per-edge scatter of avg_a * msg[src] into hneigh[dst]
    k_edge_scatter<<<(unsigned)((E + 7) / 8), 256, 0, stream>>>(
        src, dst, pp, acc, msg16, ba1, ba2, ba3, hneigh, E);
    // 10. hn16 = hneigh / max(deg,1)   (reuse tA)
    f16* hn16 = tA;
    {
        long nt = (long)N * 32;
        k_norm<<<(unsigned)((nt + 255) / 256), 256, 0, stream>>>(hneigh, acc, hn16, N);
    }
    // 11. classifier hidden: relu(h Wc1[:128] + hneigh Wc1[128:] + bc1)  (reuse tB)
    k_gemm128<true,1><<<gblk2, 256, shm2, stream>>>(hh, Wc1at, bc1, hn16, Wc1bt, tB, nullptr, N, 1);
    // 12. output: tB Wc2 + bc2 (no relu) -> f32 d_out
    k_gemm128<false,2><<<gblk1, 256, shm1, stream>>>(tB, Wc2t, bc2, nullptr, nullptr, nullptr,
                                                     (float*)d_out, N, 0);
}
